// RNNLayer_35476429866058
// MI455X (gfx1250) — matrix operand order repaired and checked by reading, not compiled, not run on hardware
//
#include <hip/hip_runtime.h>
#include <hip/hip_bf16.h>
#include <math.h>

// ---------------------------------------------------------------------------
// Elman RNN for MI455X (gfx1250, wave32, WMMA).
//   states[t] = tanh(concat(x_t, states[t-1]) @ W^T + b)
// T=512, B=64, V=1024, Hd=1024.
//
// Pipeline (ws permitting):
//   K0: zero barrier, H -> bf16 state buffer.
//   K1: inputs fp32 -> bf16 (streaming, nontemporal).
//   K2: xproj = Xbf @ Wx^T + b. bf16 WMMA GEMM, M=32768,N=1024,K=1024.
//       Wave = 1 M-tile x 4 N-tiles: one A fragment feeds 4 WMMAs; the four
//       B fragments are loaded into distinct registers before the WMMAs to
//       avoid WMMA->VALU WAR hazard NOPs (ISA 7.12.1).
//   K3: persistent serial recurrence, 64 WGs x 128 thr. WG owns 16 hidden
//       cols; 16x1024 bf16 Wh slice in LDS. Per step: 32x
//       v_wmma_f32_16x16x32_bf16 per 16x16 tile, tanh, fp32 states out
//       (nontemporal) + bf16 ping-pong state (L2-resident), grid barrier.
// Fallback (small ws): K3 fuses the x-projection (K=2048, fp32 cvt on load).
// ---------------------------------------------------------------------------

typedef __attribute__((ext_vector_type(16))) __bf16    v16bf;
typedef __attribute__((ext_vector_type(8)))  float     v8f;
typedef __attribute__((ext_vector_type(4)))  float     f32x4;   // true vector types
typedef __attribute__((ext_vector_type(4)))  unsigned  u32x4;   // for nontemporal ops

#define T_STEPS 512
#define BATCH   64
#define VDIM    1024
#define HDIM    1024
#define WCOLS   2048   // V + Hd
#define RNN_WGS 64     // one WG per 16 hidden columns (serial kernel)

// Assemble a 16-element bf16 fragment (8 VGPRs) from two 16-byte chunks.
__device__ __forceinline__ v16bf make_frag(uint4 lo, uint4 hi) {
  union { uint4 u[2]; v16bf v; } x;
  x.u[0] = lo; x.u[1] = hi;
  return x.v;
}

// A-fragment (16x32 bf16, ISA 7.12.2): lane M = lane&15, g = lane>>4.
// element e -> k = 16g+e: one contiguous 32B run (matches B fragment).
__device__ __forceinline__ v16bf a_frag_from_bf16(const __bf16* p, int g) {
  uint4 lo = *(const uint4*)(p + 16 * g);
  uint4 hi = *(const uint4*)(p + 16 * g + 8);
  return make_frag(lo, hi);
}

// Same fragment built from an fp32 row (fused fallback only).
__device__ __forceinline__ v16bf a_frag_from_f32(const float* p, int g) {
  float4 c0a = *(const float4*)(p + 16 * g);
  float4 c0b = *(const float4*)(p + 16 * g + 4);
  float4 c1a = *(const float4*)(p + 16 * g + 8);
  float4 c1b = *(const float4*)(p + 16 * g + 12);
  v16bf v;
  v[0]  = (__bf16)c0a.x; v[1]  = (__bf16)c0a.y; v[2]  = (__bf16)c0a.z; v[3]  = (__bf16)c0a.w;
  v[4]  = (__bf16)c0b.x; v[5]  = (__bf16)c0b.y; v[6]  = (__bf16)c0b.z; v[7]  = (__bf16)c0b.w;
  v[8]  = (__bf16)c1a.x; v[9]  = (__bf16)c1a.y; v[10] = (__bf16)c1a.z; v[11] = (__bf16)c1a.w;
  v[12] = (__bf16)c1b.x; v[13] = (__bf16)c1b.y; v[14] = (__bf16)c1b.z; v[15] = (__bf16)c1b.w;
  return v;
}

// B-fragment (32x16 bf16): lane n = lane&15, g = lane>>4; element e -> k = 16g+e.
// wp points at column n, k-offset (kc + 16g): one contiguous 32B run in LDS.
__device__ __forceinline__ v16bf b_frag_from_lds(const __bf16* wp) {
  uint4 lo = *(const uint4*)(wp);
  uint4 hi = *(const uint4*)(wp + 8);
  return make_frag(lo, hi);
}

// Grid-wide generation barrier (release/acquire) in workspace.
__device__ __forceinline__ void grid_sync(unsigned* bar, unsigned nwg) {
  __syncthreads();
  if (threadIdx.x == 0) {
    __threadfence();
    unsigned gen = __atomic_load_n(bar + 1, __ATOMIC_ACQUIRE);
    unsigned arr = __atomic_add_fetch(bar, 1u, __ATOMIC_ACQ_REL);
    if (arr == nwg) {
      __atomic_store_n(bar, 0u, __ATOMIC_RELAXED);
      __atomic_store_n(bar + 1, gen + 1u, __ATOMIC_RELEASE);
    } else {
      while (__atomic_load_n(bar + 1, __ATOMIC_ACQUIRE) == gen) {
        __builtin_amdgcn_s_sleep(2);
      }
    }
  }
  __syncthreads();
}

// --------------------------- K0: bar=0, H -> bf16 state[buf0] ---------------
__global__ void rnn_init_kernel(const float* __restrict__ H,
                                __bf16* __restrict__ state_bf,
                                unsigned* __restrict__ bar) {
  int i = blockIdx.x * blockDim.x + threadIdx.x;
  if (i < 2) bar[i] = 0u;
  if (i < BATCH * HDIM) state_bf[i] = (__bf16)H[i];
}

// --------------------------- K1: inputs fp32 -> bf16 ------------------------
// 8 elems/thread: 2x nontemporal b128 fp32 loads -> 1x nontemporal b128 store.
__global__ void cvt_inputs_kernel(const float* __restrict__ in,
                                  __bf16* __restrict__ outb) {
  size_t i = ((size_t)blockIdx.x * blockDim.x + threadIdx.x) * 8;
  f32x4 a = __builtin_nontemporal_load((const f32x4*)(in + i));
  f32x4 b = __builtin_nontemporal_load((const f32x4*)(in + i) + 1);
  union { __bf16 h[8]; u32x4 u; } pk;
  pk.h[0] = (__bf16)a.x; pk.h[1] = (__bf16)a.y; pk.h[2] = (__bf16)a.z; pk.h[3] = (__bf16)a.w;
  pk.h[4] = (__bf16)b.x; pk.h[5] = (__bf16)b.y; pk.h[6] = (__bf16)b.z; pk.h[7] = (__bf16)b.w;
  __builtin_nontemporal_store(pk.u, (u32x4*)(outb + i));
}

// --------------------------- K2: xproj = Xbf @ Wx^T + b ---------------------
// grid = (HDIM/64, (T*B/16)/8), block = 256 (8 waves).
// Wave = 1 M-tile x 4 N-tiles: one A fragment -> 4 WMMAs.
__global__ void xproj_kernel(const __bf16* __restrict__ xbf,
                             const float* __restrict__ W,
                             const float* __restrict__ bias,
                             float* __restrict__ xproj) {
  extern __shared__ __bf16 Wl[];               // [64][1024+8] bf16, ~132 KB
  const int PITCH = VDIM + 8;
  const int n0 = blockIdx.x * 64;

  // Cooperative fp32->bf16 load of the 64-column Wx slice (once per WG).
  for (int idx = threadIdx.x; idx < 64 * VDIM; idx += blockDim.x) {
    int r = idx >> 10, c = idx & (VDIM - 1);
    Wl[r * PITCH + c] = (__bf16)W[(size_t)(n0 + r) * WCOLS + c];
  }
  __syncthreads();

  const int wave  = threadIdx.x >> 5;
  const int lane  = threadIdx.x & 31;
  const int g     = lane >> 4;
  const int nlane = lane & 15;
  const int mtile = blockIdx.y * 8 + wave;
  const int gm    = mtile * 16 + (lane & 15);        // flat row in [T*B)

  const __bf16* xr = xbf + (size_t)gm * VDIM;

  // Per-j LDS row pointers hoisted out of the K loop.
  const __bf16* w0 = Wl + (size_t)(nlane)      * PITCH + 16 * g;
  const __bf16* w1 = Wl + (size_t)(16 + nlane) * PITCH + 16 * g;
  const __bf16* w2 = Wl + (size_t)(32 + nlane) * PITCH + 16 * g;
  const __bf16* w3 = Wl + (size_t)(48 + nlane) * PITCH + 16 * g;

  v8f acc[4];
#pragma unroll
  for (int j = 0; j < 4; ++j) {
    float bv = bias[n0 + 16 * j + nlane];
#pragma unroll
    for (int r = 0; r < 8; ++r) acc[j][r] = bv;
  }

  for (int kc = 0; kc < VDIM; kc += 32) {
    // Load A and all four B fragments into distinct registers first, so the
    // following WMMAs have no WAR hazard on the B operands (no v_nop slots).
    v16bf a  = a_frag_from_bf16(xr + kc, g);         // 2x global_load_b128
    v16bf b0 = b_frag_from_lds(w0 + kc);
    v16bf b1 = b_frag_from_lds(w1 + kc);
    v16bf b2 = b_frag_from_lds(w2 + kc);
    v16bf b3 = b_frag_from_lds(w3 + kc);
    acc[0] = __builtin_amdgcn_wmma_f32_16x16x32_bf16(false, a, false, b0,
                                                     (short)0, acc[0], false, false);
    acc[1] = __builtin_amdgcn_wmma_f32_16x16x32_bf16(false, a, false, b1,
                                                     (short)0, acc[1], false, false);
    acc[2] = __builtin_amdgcn_wmma_f32_16x16x32_bf16(false, a, false, b2,
                                                     (short)0, acc[2], false, false);
    acc[3] = __builtin_amdgcn_wmma_f32_16x16x32_bf16(false, a, false, b3,
                                                     (short)0, acc[3], false, false);
  }

#pragma unroll
  for (int j = 0; j < 4; ++j)
#pragma unroll
    for (int r = 0; r < 8; ++r) {
      int row = mtile * 16 + r + 8 * g;              // C layout: M = r + 8g
      xproj[(size_t)row * HDIM + n0 + 16 * j + nlane] = acc[j][r];
    }
}

// --------------------------- K3: serial recurrence --------------------------
// grid = 64, block = 128 (4 waves = 4 batch tiles of 16). Persistent.
__global__ void rnn_recur_kernel(const float* __restrict__ inputs,
                                 const float* __restrict__ W,
                                 const float* __restrict__ bias,
                                 const float* __restrict__ xproj,
                                 __bf16* __restrict__ state_bf,    // 2 x [64][1024]
                                 unsigned* __restrict__ bar,
                                 float* __restrict__ out,
                                 int use_xproj) {
  extern __shared__ __bf16 Wl[];                 // [16][cols+8]
  const int cols    = use_xproj ? HDIM : WCOLS;  // W_h only, or W_x|W_h fused
  const int colbase = use_xproj ? VDIM : 0;
  const int sh      = use_xproj ? 10 : 11;
  const int PITCH   = cols + 8;
  const int n0      = blockIdx.x * 16;

  for (int idx = threadIdx.x; idx < 16 * cols; idx += blockDim.x) {
    int r = idx >> sh, c = idx & (cols - 1);
    Wl[r * PITCH + c] = (__bf16)W[(size_t)(n0 + r) * WCOLS + colbase + c];
  }
  __syncthreads();

  const int wave  = threadIdx.x >> 5;            // batch tile 0..3
  const int lane  = threadIdx.x & 31;
  const int g     = lane >> 4;
  const int nlane = lane & 15;
  const int arow  = wave * 16 + (lane & 15);     // batch row for A loads
  const __bf16* wrow  = Wl + (size_t)nlane * PITCH;
  const int     wbase = use_xproj ? 0 : VDIM;    // W_h offset inside LDS slice

  for (int t = 0; t < T_STEPS; ++t) {
    const __bf16* sb  = state_bf + (size_t)(t & 1) * BATCH * HDIM;
    __bf16*       sbn = state_bf + (size_t)((t + 1) & 1) * BATCH * HDIM;

    v8f acc;
    if (use_xproj) {
      const float* xp = xproj + (size_t)t * BATCH * HDIM;
#pragma unroll
      for (int r = 0; r < 8; ++r)
        acc[r] = xp[(size_t)(wave * 16 + r + 8 * g) * HDIM + n0 + nlane];
      if (t + 1 < T_STEPS)   // warm L2/WGP$ for next step's tile
        __builtin_prefetch(xp + (size_t)BATCH * HDIM +
                           (size_t)(wave * 16 + (lane & 15)) * HDIM + n0, 0, 0);
    } else {
      const float bv = bias[n0 + nlane];
#pragma unroll
      for (int r = 0; r < 8; ++r) acc[r] = bv;
      // fused x_t @ Wx^T
      const float* xr = inputs + ((size_t)t * BATCH + arow) * VDIM;
      for (int kc = 0; kc < VDIM; kc += 32) {
        v16bf a = a_frag_from_f32(xr + kc, g);
        v16bf b = b_frag_from_lds(wrow + kc + 16 * g);
        acc = __builtin_amdgcn_wmma_f32_16x16x32_bf16(false, a, false, b,
                                                      (short)0, acc, false, false);
      }
    }

    // state_{t-1} @ Wh^T  (K = 1024, 32 WMMAs)
    const __bf16* sr = sb + (size_t)arow * HDIM;
    for (int kc = 0; kc < HDIM; kc += 32) {
      v16bf a = a_frag_from_bf16(sr + kc, g);
      v16bf b = b_frag_from_lds(wrow + wbase + kc + 16 * g);
      acc = __builtin_amdgcn_wmma_f32_16x16x32_bf16(false, a, false, b,
                                                    (short)0, acc, false, false);
    }

    // tanh, emit fp32 state (nontemporal; never re-read) + bf16 ping-pong
#pragma unroll
    for (int r = 0; r < 8; ++r) {
      float v = tanhf(acc[r]);
      int brow = wave * 16 + r + 8 * g;
      size_t o = ((size_t)t * BATCH + brow) * HDIM + n0 + nlane;
      __builtin_nontemporal_store(v, out + o);
      sbn[(size_t)brow * HDIM + n0 + nlane] = (__bf16)v;
      if (t == T_STEPS - 1)
        out[(size_t)T_STEPS * BATCH * HDIM + (size_t)brow * HDIM + n0 + nlane] = v;
    }

    if (t + 1 < T_STEPS) grid_sync(bar, RNN_WGS);
  }
}

// ---------------------------------------------------------------------------
extern "C" void kernel_launch(void* const* d_in, const int* in_sizes, int n_in,
                              void* d_out, int out_size, void* d_ws, size_t ws_size,
                              hipStream_t stream) {
  const float* inputs = (const float*)d_in[0];   // [512,64,1024] f32
  const float* H      = (const float*)d_in[1];   // [1,64,1024]   f32
  const float* W      = (const float*)d_in[2];   // [1024,2048]   f32
  const float* bias   = (const float*)d_in[3];   // [1024]        f32
  float* out = (float*)d_out;                    // states(512*64*1024) + last(64*1024)

  // ws layout: [bar 2xu32 | pad 256B] [state_bf 2x64x1024 bf16] [xproj f32] [xbf bf16]
  char* ws = (char*)d_ws;
  unsigned* bar      = (unsigned*)ws;
  __bf16*   state_bf = (__bf16*)(ws + 256);
  const size_t state_bytes = (size_t)2 * BATCH * HDIM * sizeof(__bf16);
  float* xproj = (float*)(ws + 256 + state_bytes);
  const size_t xproj_bytes = (size_t)T_STEPS * BATCH * HDIM * sizeof(float);
  __bf16* xbf = (__bf16*)(ws + 256 + state_bytes + xproj_bytes);
  const size_t xbf_bytes = (size_t)T_STEPS * BATCH * VDIM * sizeof(__bf16);
  const int use_xproj =
      (ws_size >= 256 + state_bytes + xproj_bytes + xbf_bytes) ? 1 : 0;

  // K0: zero barrier, H -> bf16 state buffer 0.
  rnn_init_kernel<<<(BATCH * HDIM + 255) / 256, 256, 0, stream>>>(H, state_bf, bar);

  if (use_xproj) {
    // K1: one-shot streaming convert of inputs to bf16.
    size_t nelem = (size_t)T_STEPS * BATCH * VDIM;           // 2^25
    cvt_inputs_kernel<<<(unsigned)(nelem / 8 / 256), 256, 0, stream>>>(inputs, xbf);

    // K2: parallel input projection (half the FLOPs, memory-bound).
    dim3 grid(HDIM / 64, (T_STEPS * BATCH / 16) / 8);        // (16, 256)
    size_t lds = (size_t)64 * (VDIM + 8) * sizeof(__bf16);   // ~132 KB
    xproj_kernel<<<grid, 256, lds, stream>>>(xbf, W, bias, xproj);
  }

  // K3: persistent serial recurrence, grid barrier per step.
  {
    size_t lds = (size_t)16 * ((use_xproj ? HDIM : WCOLS) + 8) * sizeof(__bf16);
    rnn_recur_kernel<<<RNN_WGS, 128, lds, stream>>>(inputs, W, bias, xproj,
                                                    state_bf, bar, out, use_xproj);
  }
}
